// StreamingWav2Vec2Attention_9457517985889
// MI455X (gfx1250) — matrix operand order repaired and checked by reading, not compiled, not run on hardware
//
#include <hip/hip_runtime.h>

typedef unsigned short ushort_t;
typedef __attribute__((ext_vector_type(16))) __bf16    v16bf;
typedef __attribute__((ext_vector_type(8)))  float     v8f;
typedef __attribute__((ext_vector_type(8)))  unsigned short ushort8;
typedef __attribute__((ext_vector_type(4)))  unsigned int v4u;
typedef __attribute__((ext_vector_type(8)))  int       v8i;
typedef __attribute__((ext_vector_type(4)))  int       v4i;

#define EMBED   1024
#define HEADS   16
#define HDIM    64
#define SEQ     2048
#define BATCH   2
#define MTOT    (BATCH * SEQ)   /* 4096 */
#define CHUNK   10
#define LEFTC   25
#define NEGINF  (-1e30f)

#define TILE_M  128
#define TILE_N  128
#define TILE_K  32
#define LDSTR   40              /* LDS row stride in bf16 elems; 80B rows */

#if defined(__AMDGCN__) && __has_builtin(__builtin_amdgcn_tensor_load_to_lds)
#define USE_TDM 1
#else
#define USE_TDM 0
#endif

__device__ __forceinline__ ushort_t f2bf(float f) {
    unsigned u = __float_as_uint(f);
    unsigned r = (u + 0x7FFFu + ((u >> 16) & 1u)) >> 16;   // round-to-nearest-even
    return (ushort_t)r;
}

// ---------------------------------------------------------------- fp32 -> bf16
__global__ void cvt_bf16_kernel(const float* __restrict__ in,
                                ushort_t* __restrict__ out, int n) {
    int i = blockIdx.x * blockDim.x + threadIdx.x;
    int stride = gridDim.x * blockDim.x;
    for (; i < n; i += stride) out[i] = f2bf(in[i]);
}

#if USE_TDM
// Issue a TDM load of a 2D tile (TILE_K x 128 rows of bf16) from a row-major
// [rows, EMBED] bf16 tensor into LDS, padding each 64B row with 16B so LDS row
// stride = LDSTR(40) elems. D# packing per CDNA5 ISA.
// This toolchain declares the 6-arg builtin (extra int32x8 group, zero-filled).
__device__ __forceinline__ void tdm_load_tile(const ushort_t* gtile,
                                              unsigned ldsByteAddr,
                                              unsigned rows) {
    unsigned long long ga = (unsigned long long)(size_t)gtile;
    v4u g0;
    g0.x = 1u;                                          // count=1, user descriptor
    g0.y = ldsByteAddr;                                 // lds_addr
    g0.z = (unsigned)(ga & 0xFFFFFFFFu);                // global_addr[31:0]
    g0.w = (unsigned)((ga >> 32) & 0x01FFFFFFu)         // global_addr[56:32]
         | (2u << 30);                                  // type=2 (image)
    v8i g1;
    // data_size=1(2B) | pad_enable | pad_interval=3(16 DW) | pad_amount=3(4 DW)
    g1[0] = (int)((1u << 16) | (1u << 20) | (3u << 22) | (3u << 25));
    g1[1] = (int)(((unsigned)EMBED & 0xFFFFu) << 16);   // tensor_dim0[15:0] @ bits63:48
    g1[2] = (int)(((unsigned)EMBED >> 16) | ((rows & 0xFFFFu) << 16)); // dim0 hi, dim1 lo
    g1[3] = (int)((rows >> 16) | ((unsigned)TILE_K << 16));            // dim1 hi, tile_dim0
    g1[4] = (int)TILE_M;                                // tile_dim1=128, tile_dim2=0
    g1[5] = (int)EMBED;                                 // tensor_dim0_stride low32
    g1[6] = 0;                                          // stride hi, dim1_stride lo
    g1[7] = 0;
    v4i gz;  gz[0] = gz[1] = gz[2] = gz[3] = 0;         // groups 2/3 unused (2D)
    v8i gz8; gz8[0] = gz8[1] = gz8[2] = gz8[3] = gz8[4] = gz8[5] = gz8[6] = gz8[7] = 0;
    __builtin_amdgcn_tensor_load_to_lds(g0, g1, gz, gz, gz8, 0);
}
#endif

// one K-step of the 64x32 per-wave patch: build fragments from LDS, 8 WMMAs
__device__ __forceinline__ void mma_step(const ushort_t* Ab, const ushort_t* Bb,
                                         int wm, int wn, int lrow, int half,
                                         v8f acc[4][2]) {
    // A fragment 16x32 bf16: lane = M mod 16; half selects K interleave:
    //   reg r<4 : K = 2r + 8*half ; reg r>=4 : K = 16 + 2(r-4) + 8*half
    union AF { v16bf v; unsigned u[8]; } af[4];
    #pragma unroll
    for (int mi = 0; mi < 4; ++mi) {
        const ushort_t* base = &Ab[(wm * 64 + mi * 16 + lrow) * LDSTR];
        #pragma unroll
        for (int r = 0; r < 8; ++r) {
            int kk = (r < 4) ? (2 * r + 8 * half) : (16 + 2 * (r - 4) + 8 * half);
            af[mi].u[r] = *reinterpret_cast<const unsigned*>(base + kk);
        }
    }
    // B fragment 32x16 bf16: lane = N mod 16; same K interleave as A
    union BF { v16bf v; unsigned u[8]; } bfr[2];
    #pragma unroll
    for (int ni = 0; ni < 2; ++ni) {
        const ushort_t* base = &Bb[(wn * 32 + ni * 16 + lrow) * LDSTR];
        #pragma unroll
        for (int r = 0; r < 8; ++r)
            bfr[ni].u[r] = *reinterpret_cast<const unsigned*>(base + ((r < 4) ? (2 * r + 8 * half) : (16 + 2 * (r - 4) + 8 * half)));
    }
    #pragma unroll
    for (int mi = 0; mi < 4; ++mi)
        #pragma unroll
        for (int ni = 0; ni < 2; ++ni)
            acc[mi][ni] = __builtin_amdgcn_wmma_f32_16x16x32_bf16(
                false, af[mi].v, false, bfr[ni].v,
                (short)0, acc[mi][ni], false, false);
}

// ---------------------------------------------------------------- WMMA GEMM
// Y[m,n] = (sum_k A[m,k] * W[n,k] + bias[n]) * scale
// mode 0: blockIdx.z selects (W,bias,out); out fp32 [B,H,S,D]; z==0 scaled 0.125
// mode 1: single matrix; out fp32 [M,N] row major
__global__ __launch_bounds__(256) void gemm_bf16_kernel(
    const ushort_t* __restrict__ A,
    const ushort_t* __restrict__ W0, const ushort_t* __restrict__ W1,
    const ushort_t* __restrict__ W2,
    const float* __restrict__ b0, const float* __restrict__ b1,
    const float* __restrict__ b2,
    float* __restrict__ o0, float* __restrict__ o1, float* __restrict__ o2,
    int mode)
{
    const int z = blockIdx.z;
    const ushort_t* W    = (z == 0) ? W0 : (z == 1) ? W1 : W2;
    const float*    bias = (z == 0) ? b0 : (z == 1) ? b1 : b2;
    float*          out  = (z == 0) ? o0 : (z == 1) ? o1 : o2;
    const float scale = (mode == 0 && z == 0) ? 0.125f : 1.0f;   // 64^-0.5

    const int mBase = blockIdx.y * TILE_M;
    const int nBase = blockIdx.x * TILE_N;

    const int tid  = threadIdx.x;
    const int lane = tid & 31;
    const int wave = tid >> 5;
    const int wm   = wave >> 2;      // 0..1 : 64 rows
    const int wn   = wave & 3;       // 0..3 : 32 cols
    const int lrow = lane & 15;
    const int half = lane >> 4;

    v8f acc[4][2];
    #pragma unroll
    for (int mi = 0; mi < 4; ++mi)
        #pragma unroll
        for (int ni = 0; ni < 2; ++ni)
            #pragma unroll
            for (int e = 0; e < 8; ++e) acc[mi][ni][e] = 0.0f;

#if USE_TDM
    __shared__ ushort_t As[2][TILE_M * LDSTR];
    __shared__ ushort_t Bs[2][TILE_N * LDSTR];
    const unsigned asAddr0 = (unsigned)(size_t)(void*)&As[0][0];
    const unsigned asAddr1 = (unsigned)(size_t)(void*)&As[1][0];
    const unsigned bsAddr0 = (unsigned)(size_t)(void*)&Bs[0][0];
    const unsigned bsAddr1 = (unsigned)(size_t)(void*)&Bs[1][0];

    if (wave == 0) {   // one wave drives the Tensor Data Mover
        tdm_load_tile(&A[(size_t)mBase * EMBED], asAddr0, MTOT);
        tdm_load_tile(&W[(size_t)nBase * EMBED], bsAddr0, EMBED);
    }
    const int NK = EMBED / TILE_K;
    for (int kt = 0; kt < NK; ++kt) {
        const int p = kt & 1;
        if (wave == 0) {
            if (kt + 1 < NK) {   // prefetch next tile pair into the other buffer
                const int k1 = (kt + 1) * TILE_K;
                tdm_load_tile(&A[(size_t)mBase * EMBED + k1], p ? asAddr0 : asAddr1, MTOT);
                tdm_load_tile(&W[(size_t)nBase * EMBED + k1], p ? bsAddr0 : bsAddr1, EMBED);
                // TDM ops complete in-order per wave: <=2 outstanding means the
                // current tile pair has landed in LDS.
                __builtin_amdgcn_s_wait_tensorcnt(2);
            } else {
                __builtin_amdgcn_s_wait_tensorcnt(0);
            }
        }
        __syncthreads();
        mma_step(&As[p][0], &Bs[p][0], wm, wn, lrow, half, acc);
        __syncthreads();
    }
#else
    __shared__ ushort_t As[TILE_M * LDSTR];
    __shared__ ushort_t Bs[TILE_N * LDSTR];
    const int lr = tid >> 1;         // 0..127 row in tile
    const int lc = (tid & 1) * 16;   // col 0 or 16 within K-tile
    for (int k0 = 0; k0 < EMBED; k0 += TILE_K) {
        __syncthreads();
        {
            const ushort8* ga = reinterpret_cast<const ushort8*>(
                &A[(size_t)(mBase + lr) * EMBED + k0 + lc]);
            ushort8* sa = reinterpret_cast<ushort8*>(&As[lr * LDSTR + lc]);
            sa[0] = ga[0];
            sa[1] = ga[1];
            const ushort8* gb = reinterpret_cast<const ushort8*>(
                &W[(size_t)(nBase + lr) * EMBED + k0 + lc]);
            ushort8* sb = reinterpret_cast<ushort8*>(&Bs[lr * LDSTR + lc]);
            sb[0] = gb[0];
            sb[1] = gb[1];
        }
        __syncthreads();
        mma_step(As, Bs, wm, wn, lrow, half, acc);
    }
#endif

    // C/D layout: lane = N mod 16; reg r holds M = r + 8*(lane>>4)
    #pragma unroll
    for (int mi = 0; mi < 4; ++mi) {
        #pragma unroll
        for (int ni = 0; ni < 2; ++ni) {
            const int n = nBase + wn * 32 + ni * 16 + lrow;
            const float bb = bias[n];
            #pragma unroll
            for (int r = 0; r < 8; ++r) {
                const int m = mBase + wm * 64 + mi * 16 + r + 8 * half;
                const float val = (acc[mi][ni][r] + bb) * scale;
                if (mode == 0) {
                    const int b = m >> 11, s = m & 2047;
                    const int h = n >> 6,  d = n & 63;
                    out[(((size_t)b * HEADS + h) * SEQ + s) * HDIM + d] = val;
                } else {
                    out[(size_t)m * EMBED + n] = val;
                }
            }
        }
    }
}

// ---------------------------------------------------------------- chunked attention
// One wave per query row; block per (chunk, b*H+h).
// Window for row i in chunk [cs,ce): keys j in [max(0,cs-25), ce), mask j >= i-25.
__global__ __launch_bounds__(320) void attn_kernel(
    const float* __restrict__ q, const float* __restrict__ k,
    const float* __restrict__ v, ushort_t* __restrict__ attn_out)
{
    const int ci = blockIdx.x;
    const int bh = blockIdx.y;
    const int b = bh >> 4, h = bh & 15;
    const int cs = ci * CHUNK;
    const int ce = min(cs + CHUNK, SEQ);
    const int i = cs + threadIdx.y;
    if (i >= ce) return;
    const int lane = threadIdx.x;

    const size_t headOff = ((size_t)b * HEADS + h) * SEQ * HDIM;
    const float* qrow = q + headOff + (size_t)i * HDIM;
    const float* kb = k + headOff;
    const float* vb = v + headOff;

    const int jlo = max(0, cs - LEFTC);
    const int win = ce - jlo;        // <= 35

    const float q0 = qrow[lane];
    const float q1 = qrow[lane + 32];

    float s0 = NEGINF, s1 = NEGINF;
    for (int jj = 0; jj < win; ++jj) {
        const int j = jlo + jj;
        const float* kr = kb + (size_t)j * HDIM;
        float d = q0 * kr[lane] + q1 * kr[lane + 32];
        #pragma unroll
        for (int off = 16; off > 0; off >>= 1)
            d += __shfl_xor(d, off, 32);
        if (j < i - LEFTC) d = NEGINF;      // left-context mask (right context = 0)
        if (jj == lane)      s0 = d;
        if (jj == lane + 32) s1 = d;
    }

    float mx = fmaxf(s0, s1);
    #pragma unroll
    for (int off = 16; off > 0; off >>= 1)
        mx = fmaxf(mx, __shfl_xor(mx, off, 32));

    float p0 = (lane      < win) ? __expf(s0 - mx) : 0.0f;
    float p1 = (lane + 32 < win) ? __expf(s1 - mx) : 0.0f;
    float sum = p0 + p1;
    #pragma unroll
    for (int off = 16; off > 0; off >>= 1)
        sum += __shfl_xor(sum, off, 32);
    const float inv = 1.0f / sum;

    float o0 = 0.0f, o1 = 0.0f;
    for (int jj = 0; jj < win; ++jj) {
        const int j = jlo + jj;
        const float p = (jj < 32) ? __shfl(p0, jj, 32) : __shfl(p1, jj - 32, 32);
        const float* vr = vb + (size_t)j * HDIM;
        o0 += p * vr[lane];
        o1 += p * vr[lane + 32];
    }

    ushort_t* orow = attn_out + ((size_t)b * SEQ + i) * EMBED + h * HDIM;
    orow[lane]      = f2bf(o0 * inv);
    orow[lane + 32] = f2bf(o1 * inv);
}

// ---------------------------------------------------------------- launcher
extern "C" void kernel_launch(void* const* d_in, const int* in_sizes, int n_in,
                              void* d_out, int out_size, void* d_ws, size_t ws_size,
                              hipStream_t stream) {
    (void)in_sizes; (void)n_in; (void)out_size; (void)ws_size;

    const float* x  = (const float*)d_in[0];
    const float* Wq = (const float*)d_in[1];
    const float* bq = (const float*)d_in[2];
    const float* Wk = (const float*)d_in[3];
    const float* bk = (const float*)d_in[4];
    const float* Wv = (const float*)d_in[5];
    const float* bv = (const float*)d_in[6];
    const float* Wo = (const float*)d_in[7];
    const float* bo = (const float*)d_in[8];
    float* out = (float*)d_out;

    char* ws = (char*)d_ws;
    ushort_t* xb    = (ushort_t*)ws; ws += (size_t)MTOT * EMBED * 2;
    ushort_t* wqb   = (ushort_t*)ws; ws += (size_t)EMBED * EMBED * 2;
    ushort_t* wkb   = (ushort_t*)ws; ws += (size_t)EMBED * EMBED * 2;
    ushort_t* wvb   = (ushort_t*)ws; ws += (size_t)EMBED * EMBED * 2;
    ushort_t* wob   = (ushort_t*)ws; ws += (size_t)EMBED * EMBED * 2;
    float*    qf    = (float*)ws;    ws += (size_t)MTOT * EMBED * 4;
    float*    kf    = (float*)ws;    ws += (size_t)MTOT * EMBED * 4;
    float*    vf    = (float*)ws;    ws += (size_t)MTOT * EMBED * 4;
    ushort_t* attnb = (ushort_t*)ws; ws += (size_t)MTOT * EMBED * 2;

    const int nx = MTOT * EMBED;
    const int nw = EMBED * EMBED;
    cvt_bf16_kernel<<<(nx + 255) / 256, 256, 0, stream>>>(x,  xb,  nx);
    cvt_bf16_kernel<<<(nw + 255) / 256, 256, 0, stream>>>(Wq, wqb, nw);
    cvt_bf16_kernel<<<(nw + 255) / 256, 256, 0, stream>>>(Wk, wkb, nw);
    cvt_bf16_kernel<<<(nw + 255) / 256, 256, 0, stream>>>(Wv, wvb, nw);
    cvt_bf16_kernel<<<(nw + 255) / 256, 256, 0, stream>>>(Wo, wob, nw);

    // fused Q/K/V projections (z picks matrix); q scaled by 0.125 inside
    dim3 gq(EMBED / TILE_N, MTOT / TILE_M, 3);
    gemm_bf16_kernel<<<gq, 256, 0, stream>>>(xb, wqb, wkb, wvb,
                                             bq, bk, bv, qf, kf, vf, 0);

    dim3 ga((SEQ + CHUNK - 1) / CHUNK, BATCH * HEADS);
    dim3 ba(32, CHUNK);
    attn_kernel<<<ga, ba, 0, stream>>>(qf, kf, vf, attnb);

    // output projection straight into d_out
    dim3 go(EMBED / TILE_N, MTOT / TILE_M, 1);
    gemm_bf16_kernel<<<go, 256, 0, stream>>>(attnb, wob, wob, wob,
                                             bo, bo, bo, out, out, out, 1);
}
